// EigenAttention_721554505771
// MI455X (gfx1250) — matrix operand order repaired and checked by reading, not compiled, not run on hardware
//
#include <hip/hip_runtime.h>
#include <hip/hip_bf16.h>
#include <stdint.h>

// Problem constants (match reference)
#define B_    2
#define S_    2048
#define H_    1024
#define NH_   16
#define HD_   64
#define KR_   256
#define MROWS (B_*S_)   // 4096 token rows

typedef __attribute__((ext_vector_type(16))) __bf16 v16bf;
typedef __attribute__((ext_vector_type(8)))  float  v8f;
typedef __attribute__((ext_vector_type(4)))  unsigned int u32x4;
typedef __attribute__((ext_vector_type(8)))  int i32x8;
typedef __attribute__((ext_vector_type(4)))  int i32x4;

union Frag16 { v16bf v; uint4 q[2]; };
union U8bf   { uint4 q; __bf16 e[8]; };

// Native RNE f32 -> bf16 (hardware v_cvt, ~1 VALU op vs 3 for manual rounding)
__device__ __forceinline__ __bf16 f2bf(float f) { return (__bf16)f; }

// Pack two converted bf16 into one dword (enables b64 LDS stores)
__device__ __forceinline__ unsigned pack2bf(float a, float b) {
  union { __bf16 e[2]; unsigned u; } t;
  t.e[0] = (__bf16)a; t.e[1] = (__bf16)b;
  return t.u;
}

__device__ __forceinline__ void store_out(float* p, float v)  { *p = v; }
__device__ __forceinline__ void store_out(__bf16* p, float v) { *p = f2bf(v); }

// ---------------------------------------------------------------------------
// TDM: DMA a 64x64 bf16 tile (row stride 1024 elem) from global into LDS with
// row padding 128B data + 16B pad = 144B rows (matches __bf16[64][72]).
// Descriptor per CDNA5 ISA.
// ---------------------------------------------------------------------------
#if __has_builtin(__builtin_amdgcn_tensor_load_to_lds) && \
    __has_builtin(__builtin_amdgcn_s_wait_tensorcnt)
#define USE_TDM 1
#else
#define USE_TDM 0
#endif

#if USE_TDM
__device__ __forceinline__ void tdm_load_tile64x64_bf16(unsigned lds_addr,
                                                        unsigned long long gaddr)
{
  u32x4 g0;
  g0[0] = 1u;                                   // count=1, user descriptor
  g0[1] = lds_addr;                             // LDS byte address
  g0[2] = (unsigned)(gaddr & 0xFFFFFFFFu);      // global_addr[31:0]
  g0[3] = (unsigned)((gaddr >> 32) & 0x01FFFFFFu) | (2u << 30); // [56:32]|type=2

  i32x8 g1;
  // data_size=1 (2B), pad_enable, pad_interval=4 (32 DW=128B), pad_amount=3 (4 DW=16B)
  g1[0] = (int)((1u << 16) | (1u << 20) | (4u << 22) | (3u << 25));
  g1[1] = (int)(64u << 16);                     // tensor_dim0[15:0]=64
  g1[2] = (int)(64u << 16);                     // dim0 hi=0 | tensor_dim1[15:0]=64
  g1[3] = (int)(64u << 16);                     // dim1 hi=0 | tile_dim0=64
  g1[4] = (int)64;                              // tile_dim1=64 | tile_dim2=0
  g1[5] = (int)1024;                            // tensor_dim0_stride=1024 elem
  g1[6] = 0;                                    // stride0 hi | dim1_stride lo
  g1[7] = 0;
  i32x4 z = {0, 0, 0, 0};
#if __clang_major__ >= 23
  i32x8 z8 = {0, 0, 0, 0, 0, 0, 0, 0};
  __builtin_amdgcn_tensor_load_to_lds(g0, g1, z, z, z8, 0);
#else
  __builtin_amdgcn_tensor_load_to_lds(g0, g1, z, z, 0);
#endif
}
#endif

// ---------------------------------------------------------------------------
// Generic WMMA GEMM: C[M,N] = (A - a_sub) @ op(Bw) + bias  (C in OutT)
//   TRANS_B=true : op(Bw) = Bw^T, Bw is [N, Kd] row-major
//   TRANS_B=false: op(Bw) = Bw,   Bw is [Kd, N] row-major
// Tiles: 128x128x32, 256 threads = 8 waves (2x4); wave tile 64x32 (4x2 wmma)
// ---------------------------------------------------------------------------
template<bool TRANS_B, typename OutT>
__global__ __launch_bounds__(256)
void wmma_gemm(const float* __restrict__ A, const float* __restrict__ Bw,
               const float* __restrict__ bias, const float* __restrict__ a_sub,
               OutT* __restrict__ C,
               int M, int N, int Kd, int lda, int ldb, int ldc)
{
  __shared__ __align__(16) __bf16 As[128][40];   // [m][k], 80B rows
  __shared__ __align__(16) __bf16 Bs[128][40];   // [n][k]

  const int tid    = threadIdx.x;
  const int wid    = tid >> 5;
  const int lane   = tid & 31;
  const int hlf    = lane >> 4;
  const int ln     = lane & 15;
  const int wave_m = wid >> 2;      // 0..1
  const int wave_n = wid & 3;       // 0..3

  const int tile_n = blockIdx.x * 128;
  const int tile_m = blockIdx.y * 128;

  v8f acc[4][2];
  #pragma unroll
  for (int i = 0; i < 4; ++i)
    #pragma unroll
    for (int j = 0; j < 2; ++j) acc[i][j] = {};

  for (int kk = 0; kk < Kd; kk += 32) {
    __syncthreads();
    #pragma unroll
    for (int p = 0; p < 4; ++p) {
      int idx = p * 1024 + tid * 4;
      int r = idx >> 5, c = idx & 31;
      const float* g = &A[(size_t)(tile_m + r) * lda + kk + c];
      if (kk + 32 < Kd) __builtin_prefetch(g + 32, 0, 3);
      float4 f = *(const float4*)g;
      if (a_sub) {
        f.x -= a_sub[kk + c + 0]; f.y -= a_sub[kk + c + 1];
        f.z -= a_sub[kk + c + 2]; f.w -= a_sub[kk + c + 3];
      }
      uint2 pk; pk.x = pack2bf(f.x, f.y); pk.y = pack2bf(f.z, f.w);
      *(uint2*)&As[r][c] = pk;                   // one ds_store_b64
    }
    if (TRANS_B) {
      #pragma unroll
      for (int p = 0; p < 4; ++p) {
        int idx = p * 1024 + tid * 4;
        int r = idx >> 5, c = idx & 31;
        const float* g = &Bw[(size_t)(tile_n + r) * ldb + kk + c];
        if (kk + 32 < Kd) __builtin_prefetch(g + 32, 0, 3);
        float4 f = *(const float4*)g;
        uint2 pk; pk.x = pack2bf(f.x, f.y); pk.y = pack2bf(f.z, f.w);
        *(uint2*)&Bs[r][c] = pk;
      }
    } else {
      #pragma unroll
      for (int p = 0; p < 4; ++p) {
        int idx = p * 1024 + tid * 4;
        int k = idx >> 7, n = idx & 127;     // transpose on store
        float4 f = *(const float4*)&Bw[(size_t)(kk + k) * ldb + tile_n + n];
        Bs[n + 0][k] = f2bf(f.x); Bs[n + 1][k] = f2bf(f.y);
        Bs[n + 2][k] = f2bf(f.z); Bs[n + 3][k] = f2bf(f.w);
      }
    }
    __syncthreads();

    Frag16 a[4], b[2];
    #pragma unroll
    for (int i = 0; i < 4; ++i) {
      int m = wave_m * 64 + i * 16 + ln;
      a[i].q[0] = *(const uint4*)&As[m][8 * hlf];
      a[i].q[1] = *(const uint4*)&As[m][16 + 8 * hlf];
    }
    #pragma unroll
    for (int j = 0; j < 2; ++j) {
      int n = wave_n * 32 + j * 16 + ln;
      b[j].q[0] = *(const uint4*)&Bs[n][8 * hlf];
      b[j].q[1] = *(const uint4*)&Bs[n][16 + 8 * hlf];
    }
    #pragma unroll
    for (int i = 0; i < 4; ++i)
      #pragma unroll
      for (int j = 0; j < 2; ++j)
        acc[i][j] = __builtin_amdgcn_wmma_f32_16x16x32_bf16(
            false, a[i].v, false, b[j].v, (short)0, acc[i][j], false, false);
  }

  #pragma unroll
  for (int j = 0; j < 2; ++j) {
    int col = tile_n + wave_n * 32 + j * 16 + ln;
    float bv = bias ? bias[col] : 0.f;
    #pragma unroll
    for (int i = 0; i < 4; ++i) {
      int rowbase = tile_m + wave_m * 64 + i * 16 + 8 * hlf;
      #pragma unroll
      for (int r = 0; r < 8; ++r)
        store_out(&C[(size_t)(rowbase + r) * ldc + col], acc[i][j][r] + bv);
    }
  }
}

// ---------------------------------------------------------------------------
// Fused causal flash-attention. K/V are bf16 [4096,1024]. 64 q-rows per WG,
// 4 waves x 16 rows. K tile staged by the Tensor Data Mover (TDM); V staged
// manually transposed. Causal mask only on the diagonal block (uniform branch).
// ---------------------------------------------------------------------------
__global__ __launch_bounds__(128)
void flash_attn(const float* __restrict__ qkv, const __bf16* __restrict__ kfull,
                const __bf16* __restrict__ vfull, float* __restrict__ out)
{
  __shared__ __align__(16) __bf16 Qs[64][72];   // [q][d] (pre-scaled)
  __shared__ __align__(16) __bf16 Ks[64][72];   // [kpos][d]  (TDM target)
  __shared__ __align__(16) __bf16 Vs[64][72];   // transposed: [d][kpos]
  __shared__ __align__(16) __bf16 Ps[64][72];   // [q][kpos] probabilities

  const int tid  = threadIdx.x;
  const int wid  = tid >> 5;
  const int lane = tid & 31;
  const int hlf  = lane >> 4;
  const int ln   = lane & 15;

  const int bq = blockIdx.x;
  const int h  = blockIdx.y;
  const int b  = blockIdx.z;
  const float scale = 0.125f;  // 1/sqrt(64)

  // ---- stage Q (scaled, f32 -> bf16, packed b64 stores) ----
  #pragma unroll
  for (int p = 0; p < 8; ++p) {
    int idx = p * 512 + tid * 4;
    int r = idx >> 6, d = idx & 63;
    int s = bq * 64 + r;
    const float4 f = *(const float4*)&qkv[(size_t)(b * S_ + s) * (3 * H_) + h * HD_ + d];
    uint2 pk;
    pk.x = pack2bf(f.x * scale, f.y * scale);
    pk.y = pack2bf(f.z * scale, f.w * scale);
    *(uint2*)&Qs[r][d] = pk;
  }
  __syncthreads();

  Frag16 aq[2];
  {
    int m = wid * 16 + ln;
    #pragma unroll
    for (int c = 0; c < 2; ++c) {
      aq[c].q[0] = *(const uint4*)&Qs[m][32 * c + 8 * hlf];
      aq[c].q[1] = *(const uint4*)&Qs[m][32 * c + 16 + 8 * hlf];
    }
  }

  float row_max[8], row_sum[8];
  v8f o_acc[4];
  #pragma unroll
  for (int r = 0; r < 8; ++r) { row_max[r] = -3.0e38f; row_sum[r] = 0.f; }
  #pragma unroll
  for (int j = 0; j < 4; ++j) o_acc[j] = {};

  const int qbase = bq * 64 + wid * 16;

  for (int kb = 0; kb <= bq; ++kb) {
    __syncthreads();   // previous compute done reading Ks/Vs

    // ---- K tile via TDM (wave 0 only issues the DMA) ----
#if USE_TDM
    if (wid == 0) {
      unsigned lds_addr = (unsigned)(uintptr_t)&Ks[0][0];
      unsigned long long ga = (unsigned long long)(uintptr_t)
          &kfull[(size_t)(b * S_ + kb * 64) * H_ + h * HD_];
      tdm_load_tile64x64_bf16(lds_addr, ga);
    }
#else
    #pragma unroll
    for (int p = 0; p < 4; ++p) {
      int idx = (p * 128 + tid) * 8;
      int r = idx >> 6, d = idx & 63;
      *(uint4*)&Ks[r][d] =
          *(const uint4*)&kfull[(size_t)(b * S_ + kb * 64 + r) * H_ + h * HD_ + d];
    }
#endif
    // ---- V tile: bf16 load + transpose into Vs[d][kpos] ----
    #pragma unroll
    for (int p = 0; p < 4; ++p) {
      int idx = (p * 128 + tid) * 8;
      int r = idx >> 6, d = idx & 63;
      U8bf u;
      u.q = *(const uint4*)&vfull[(size_t)(b * S_ + kb * 64 + r) * H_ + h * HD_ + d];
      #pragma unroll
      for (int j = 0; j < 8; ++j) Vs[d + j][r] = u.e[j];
    }
#if USE_TDM
    if (wid == 0) __builtin_amdgcn_s_wait_tensorcnt(0);
#endif
    __syncthreads();

    // ---- per-block compute; MASKED folds to constant after inlining ----
    auto process = [&](bool MASKED) {
      v8f s_acc[4];
      #pragma unroll
      for (int tn = 0; tn < 4; ++tn) s_acc[tn] = {};
      #pragma unroll
      for (int tn = 0; tn < 4; ++tn) {
        int n = tn * 16 + ln;
        #pragma unroll
        for (int c = 0; c < 2; ++c) {
          Frag16 bk;
          bk.q[0] = *(const uint4*)&Ks[n][32 * c + 8 * hlf];
          bk.q[1] = *(const uint4*)&Ks[n][32 * c + 16 + 8 * hlf];
          s_acc[tn] = __builtin_amdgcn_wmma_f32_16x16x32_bf16(
              false, aq[c].v, false, bk.v, (short)0, s_acc[tn], false, false);
        }
      }

      #pragma unroll
      for (int r = 0; r < 8; ++r) {
        int m = r + 8 * hlf;
        int qpos = qbase + m;
        float sv[4];
        float cur = -3.0e38f;
        #pragma unroll
        for (int tn = 0; tn < 4; ++tn) {
          float sc = s_acc[tn][r];
          if (MASKED) {
            int kpos = kb * 64 + tn * 16 + ln;
            sc = (kpos <= qpos) ? sc : -3.0e38f;
          }
          sv[tn] = sc;
          cur = fmaxf(cur, sc);
        }
        #pragma unroll
        for (int off = 1; off < 16; off <<= 1)
          cur = fmaxf(cur, __shfl_xor(cur, off, 32));
        float nm   = fmaxf(row_max[r], cur);
        float corr = __expf(row_max[r] - nm);
        row_max[r] = nm;
        float psum = 0.f;
        #pragma unroll
        for (int tn = 0; tn < 4; ++tn) {
          float pp = __expf(sv[tn] - nm);
          psum += pp;
          Ps[wid * 16 + m][tn * 16 + ln] = f2bf(pp);
        }
        #pragma unroll
        for (int off = 1; off < 16; off <<= 1)
          psum += __shfl_xor(psum, off, 32);
        row_sum[r] = row_sum[r] * corr + psum;
        #pragma unroll
        for (int j = 0; j < 4; ++j) o_acc[j][r] *= corr;
      }

      #pragma unroll
      for (int c = 0; c < 2; ++c) {
        Frag16 ap;
        int m = wid * 16 + ln;
        ap.q[0] = *(const uint4*)&Ps[m][32 * c + 8 * hlf];
        ap.q[1] = *(const uint4*)&Ps[m][32 * c + 16 + 8 * hlf];
        #pragma unroll
        for (int j = 0; j < 4; ++j) {
          Frag16 bv;
          int n = j * 16 + ln;
          bv.q[0] = *(const uint4*)&Vs[n][32 * c + 8 * hlf];
          bv.q[1] = *(const uint4*)&Vs[n][32 * c + 16 + 8 * hlf];
          o_acc[j] = __builtin_amdgcn_wmma_f32_16x16x32_bf16(
              false, ap.v, false, bv.v, (short)0, o_acc[j], false, false);
        }
      }
    };

    if (kb == bq) process(true);   // diagonal block: causal mask
    else         process(false);   // interior block: no mask, no cndmasks
  }

  #pragma unroll
  for (int r = 0; r < 8; ++r) {
    float inv = 1.f / row_sum[r];
    int sq = bq * 64 + wid * 16 + r + 8 * hlf;
    #pragma unroll
    for (int j = 0; j < 4; ++j)
      out[(size_t)(b * S_ + sq) * H_ + h * HD_ + j * 16 + ln] = o_acc[j][r] * inv;
  }
}

// ---------------------------------------------------------------------------
extern "C" void kernel_launch(void* const* d_in, const int* in_sizes, int n_in,
                              void* d_out, int out_size, void* d_ws, size_t ws_size,
                              hipStream_t stream)
{
  const float* hidden   = (const float*)d_in[0];
  const float* c_attn_w = (const float*)d_in[1];
  const float* c_attn_b = (const float*)d_in[2];
  const float* c_proj_w = (const float*)d_in[3];
  const float* c_proj_b = (const float*)d_in[4];
  const float* k_proj   = (const float*)d_in[5];
  const float* k_mean   = (const float*)d_in[6];
  const float* v_proj   = (const float*)d_in[7];
  const float* v_mean   = (const float*)d_in[8];
  float* outp = (float*)d_out;

  char* ws = (char*)d_ws;
  float*  qkv   = (float*)ws;  ws += (size_t)MROWS * 3 * H_ * sizeof(float);  // 48MB
  __bf16* kfull = (__bf16*)ws; ws += (size_t)MROWS * H_ * sizeof(__bf16);     // 8MB
  __bf16* vfull = (__bf16*)ws; ws += (size_t)MROWS * H_ * sizeof(__bf16);     // 8MB
  float*  attn  = (float*)ws;  ws += (size_t)MROWS * H_ * sizeof(float);      // 16MB
  float*  cbuf  = (float*)ws;  ws += (size_t)MROWS * KR_ * sizeof(float);     // 4MB

  dim3 blk(256);

  // 1) qkv = hidden @ c_attn_w^T + c_attn_b           [4096,3072] f32
  wmma_gemm<true, float><<<dim3(3 * H_ / 128, MROWS / 128), blk, 0, stream>>>(
      hidden, c_attn_w, c_attn_b, nullptr, qkv,
      MROWS, 3 * H_, H_, H_, H_, 3 * H_);

  // 2) ck = (k - k_mean) @ k_proj^T                   [4096,256] f32
  wmma_gemm<true, float><<<dim3(KR_ / 128, MROWS / 128), blk, 0, stream>>>(
      qkv + H_, k_proj, nullptr, k_mean, cbuf,
      MROWS, KR_, H_, 3 * H_, H_, KR_);

  // 3) kfull = ck @ k_proj + k_mean                   [4096,1024] bf16
  wmma_gemm<false, __bf16><<<dim3(H_ / 128, MROWS / 128), blk, 0, stream>>>(
      cbuf, k_proj, k_mean, nullptr, kfull,
      MROWS, H_, KR_, KR_, H_, H_);

  // 4) cv = (v - v_mean) @ v_proj^T ; vfull = cv @ v_proj + v_mean (bf16)
  wmma_gemm<true, float><<<dim3(KR_ / 128, MROWS / 128), blk, 0, stream>>>(
      qkv + 2 * H_, v_proj, nullptr, v_mean, cbuf,
      MROWS, KR_, H_, 3 * H_, H_, KR_);
  wmma_gemm<false, __bf16><<<dim3(H_ / 128, MROWS / 128), blk, 0, stream>>>(
      cbuf, v_proj, v_mean, nullptr, vfull,
      MROWS, H_, KR_, KR_, H_, H_);

  // 5) fused causal flash attention -> attn [4096,1024] ([B,S,H] layout)
  flash_attn<<<dim3(S_ / 64, NH_, B_), dim3(128), 0, stream>>>(
      qkv, kfull, vfull, attn);

  // 6) out = attn @ c_proj_w^T + c_proj_b
  wmma_gemm<true, float><<<dim3(H_ / 128, MROWS / 128), blk, 0, stream>>>(
      attn, c_proj_w, c_proj_b, nullptr, outp,
      MROWS, H_, H_, H_, H_, H_);
}
